// Attention_31344671326347
// MI455X (gfx1250) — matrix operand order repaired and checked by reading, not compiled, not run on hardware
//
#include <hip/hip_runtime.h>
#include <stdint.h>

typedef unsigned short u16;
typedef __bf16 v16bf __attribute__((ext_vector_type(16)));
typedef float  v8f   __attribute__((ext_vector_type(8)));
typedef int    v4i   __attribute__((ext_vector_type(4)));

typedef __attribute__((address_space(1))) v4i glb_v4i;
typedef __attribute__((address_space(3))) v4i lds_v4i;

__device__ __forceinline__ u16 f2bf(float f) {
  union { float f; uint32_t u; } c; c.f = f;
  uint32_t u = c.u;
  uint32_t r = u + 0x7FFFu + ((u >> 16) & 1u);   // round-to-nearest-even
  return (u16)(r >> 16);
}

// ---------------------------------------------------------------- convert
__global__ __launch_bounds__(256)
void cvt_f32_bf16(const float* __restrict__ in, u16* __restrict__ out, int n) {
  int i = (blockIdx.x * 256 + threadIdx.x) * 4;
  if (i + 3 < n) {
    float4 v = *(const float4*)(in + i);
    union { u16 h[4]; uint2 u; } o;
    o.h[0] = f2bf(v.x); o.h[1] = f2bf(v.y);
    o.h[2] = f2bf(v.z); o.h[3] = f2bf(v.w);
    *(uint2*)(out + i) = o.u;
  } else {
    for (int j = i; j < n && j < i + 4; ++j) out[j] = f2bf(in[j]);
  }
}

// ---------------------------------------------------------------- GEMM
// C[m,n] = (sum_k A[m,k]*B[n,k] + bias[n]) * scale
// A: [M,K] bf16 row-major, B: [N,K] bf16 row-major (i.e. C = A * B^T)
// OUT_F32: write float, else bf16. TRANS: store C[n*ldc + m].
// Workgroup: 128 thr = 4 waves (2x2), C tile 128x128; wave tile 64x64.
// Double-buffered LDS; staging uses CDNA5 GLOBAL_LOAD_ASYNC_TO_LDS_B128
// (ASYNCcnt-tracked DMA). All per-lane offsets are 32-bit so global
// accesses lower to SGPR-base + 32-bit VGPR offset (GVS addressing).
template <bool OUT_F32, bool TRANS>
__global__ __launch_bounds__(128, 1)
void gemm_bf16_nt(const u16* __restrict__ A, int lda, long long aBatch,
                  const u16* __restrict__ Bm, int ldb, long long bBatch,
                  void* __restrict__ Cv, int ldc, long long cBatch,
                  const float* __restrict__ bias, float scale, int K) {
  // 128 rows x 32 bf16 cols, padded to 40 (80B rows, 16B aligned), x2 buffers
  __shared__ u16 As[2][128 * 40];
  __shared__ u16 Bs[2][128 * 40];

  const int tid  = threadIdx.x;
  const int lane = tid & 31;
  const int wave = tid >> 5;             // 4 waves
  const int wm   = wave & 1;             // 2 waves along M
  const int wn   = wave >> 1;            // 2 waves along N
  const int m0   = blockIdx.y * 128;
  const int n0   = blockIdx.x * 128;
  const int z    = blockIdx.z;

  const u16* Ab = A  + (long long)z * aBatch;   // uniform (SGPR) bases
  const u16* Bb = Bm + (long long)z * bBatch;

  v8f acc[4][4];                         // [mi][ni], 64x64 wave tile
  for (int mi = 0; mi < 4; ++mi)
    for (int ni = 0; ni < 4; ++ni)
      for (int r = 0; r < 8; ++r) acc[mi][ni][r] = 0.0f;

  float bv[4];
  for (int ni = 0; ni < 4; ++ni) {
    int n = n0 + wn * 64 + ni * 16 + (lane & 15);
    bv[ni] = bias ? bias[n] : 0.0f;
  }

  const int ko   = lane >> 4;            // half-wave selector (ISA frag layout)
  const int lrow = lane & 15;

  // staging assignment: 512 x 16B chunks per 128x32 tile, 4 per thread.
  // 32-bit element offsets (all tensors < 2^31 elements).
  unsigned aOff[4], bOff[4];
  int srow[4], sseg[4];
  #pragma unroll
  for (int c = 0; c < 4; ++c) {
    int idx = tid + c * 128;
    srow[c] = idx >> 2;
    sseg[c] = idx & 3;
    aOff[c] = (unsigned)(m0 + srow[c]) * (unsigned)lda + (unsigned)(sseg[c] * 8);
    bOff[c] = (unsigned)(n0 + srow[c]) * (unsigned)ldb + (unsigned)(sseg[c] * 8);
  }

  // async DMA one 128x32 A tile + one 128x32 B tile into buffer `buf`
  auto stage_async = [&](int buf, unsigned kk) {
    #pragma unroll
    for (int c = 0; c < 4; ++c) {
      u16* la = &As[buf][srow[c] * 40 + sseg[c] * 8];
      u16* lb = &Bs[buf][srow[c] * 40 + sseg[c] * 8];
      __builtin_amdgcn_global_load_async_to_lds_b128(
          (glb_v4i*)(Ab + (aOff[c] + kk)), (lds_v4i*)la, 0, 0);
      __builtin_amdgcn_global_load_async_to_lds_b128(
          (glb_v4i*)(Bb + (bOff[c] + kk)), (lds_v4i*)lb, 0, 0);
    }
  };

  // one 16-wmma compute phase against buffer `cur`
  auto compute_tile = [&](int cur) {
    union F { v16bf v; uint4 u[2]; };
    // B frag: lane holds column N=lane%16, K = (lane/16)*16 + e (16 contiguous)
    F bfrag[4];
    #pragma unroll
    for (int ni = 0; ni < 4; ++ni) {
      int r = wn * 64 + ni * 16 + lrow;
      bfrag[ni].u[0] = *(const uint4*)&Bs[cur][r * 40 + ko * 16];
      bfrag[ni].u[1] = *(const uint4*)&Bs[cur][r * 40 + ko * 16 + 8];
    }
    // A frag: lane holds row M=lane%16, K = (lane/16)*16 + e (16 contiguous)
    #pragma unroll
    for (int mi = 0; mi < 4; ++mi) {
      F af;
      int r = wm * 64 + mi * 16 + lrow;
      af.u[0] = *(const uint4*)&As[cur][r * 40 + ko * 16];
      af.u[1] = *(const uint4*)&As[cur][r * 40 + ko * 16 + 8];
      #pragma unroll
      for (int ni = 0; ni < 4; ++ni) {
        acc[mi][ni] = __builtin_amdgcn_wmma_f32_16x16x32_bf16(
            false, af.v, false, bfrag[ni].v, (short)0, acc[mi][ni],
            false, false);
      }
    }
  };

  const int nk = K >> 5;

  // ---- prologue: async-stage tile 0 into buffer 0
  stage_async(0, 0);
  asm volatile("s_wait_asynccnt 0x0" ::: "memory");
  __syncthreads();

  // ---- steady state: async loads for tile k+1 overlap 16 wmma on tile k
  for (int kt = 0; kt + 1 < nk; ++kt) {
    const int cur = kt & 1, nxt = cur ^ 1;
    stage_async(nxt, (unsigned)(kt + 1) << 5);   // DMA into other buffer
    compute_tile(cur);                           // wmma while DMA in flight
    asm volatile("s_wait_asynccnt 0x0" ::: "memory");
    __syncthreads();
  }
  compute_tile((nk - 1) & 1);            // peeled final tile

  // epilogue: C/D layout -> M = r + 8*(lane/16), N = lane%16 per 16x16 tile.
  // Uniform batch base + per-lane 32-bit offset; single address chain
  // (TRANS is a template constant).
  float* Cf = (float*)Cv + (long long)z * cBatch;
  u16*   Ch = (u16*)Cv   + (long long)z * cBatch;
  const unsigned ldcu = (unsigned)ldc;
  for (int mi = 0; mi < 4; ++mi) {
    for (int ni = 0; ni < 4; ++ni) {
      unsigned n = (unsigned)(n0 + wn * 64 + ni * 16 + lrow);
      #pragma unroll
      for (int r = 0; r < 8; ++r) {
        unsigned m = (unsigned)(m0 + wm * 64 + mi * 16 + r + 8 * ko);
        float v = (acc[mi][ni][r] + bv[ni]) * scale;
        unsigned off = TRANS ? (n * ldcu + m) : (m * ldcu + n);
        if (OUT_F32) Cf[off] = v;
        else         Ch[off] = f2bf(v);
      }
    }
  }
}

// ---------------------------------------------------------------- softmax
// one block per row; cols == 2048 == 8 * 256
__global__ __launch_bounds__(256)
void softmax_rows(const float* __restrict__ S, u16* __restrict__ P, int cols) {
  __shared__ float red[256];
  const int row = blockIdx.x;
  const int tid = threadIdx.x;
  const float* rp = S + (long long)row * cols;
  u16* op = P + (long long)row * cols;

  float vals[8];
  float lmax = -3.4e38f;
  #pragma unroll
  for (int j = 0; j < 8; ++j) {
    vals[j] = rp[tid + j * 256];
    lmax = fmaxf(lmax, vals[j]);
  }
  red[tid] = lmax; __syncthreads();
  for (int s = 128; s > 0; s >>= 1) {
    if (tid < s) red[tid] = fmaxf(red[tid], red[tid + s]);
    __syncthreads();
  }
  float m = red[0]; __syncthreads();

  float lsum = 0.0f;
  #pragma unroll
  for (int j = 0; j < 8; ++j) { vals[j] = __expf(vals[j] - m); lsum += vals[j]; }
  red[tid] = lsum; __syncthreads();
  for (int s = 128; s > 0; s >>= 1) {
    if (tid < s) red[tid] += red[tid + s];
    __syncthreads();
  }
  float inv = 1.0f / red[0];
  #pragma unroll
  for (int j = 0; j < 8; ++j) op[tid + j * 256] = f2bf(vals[j] * inv);
}

// ---------------------------------------------------------------- launch
extern "C" void kernel_launch(void* const* d_in, const int* in_sizes, int n_in,
                              void* d_out, int out_size, void* d_ws, size_t ws_size,
                              hipStream_t stream) {
  const float* x  = (const float*)d_in[0];
  const float* Wq = (const float*)d_in[1];
  const float* bq = (const float*)d_in[2];
  const float* Wk = (const float*)d_in[3];
  const float* bk = (const float*)d_in[4];
  const float* Wv = (const float*)d_in[5];
  const float* bv = (const float*)d_in[6];
  float* out = (float*)d_out;

  const int B = 4, S = 2048, E = 1024, D = 1024;
  const long long M = (long long)B * S;   // 8192 rows

  char* ws = (char*)d_ws;
  size_t off = 0;
  auto carve = [&](size_t bytes) -> char* {
    char* p = ws + off;
    off += (bytes + 255) & ~(size_t)255;
    return p;
  };
  u16*   xb  = (u16*)carve((size_t)M * E * 2);        // x in bf16
  u16*   wqb = (u16*)carve((size_t)D * E * 2);
  u16*   wkb = (u16*)carve((size_t)D * E * 2);
  u16*   wvb = (u16*)carve((size_t)D * E * 2);
  u16*   qb  = (u16*)carve((size_t)M * D * 2);        // Q (pre-scaled 1/32)
  u16*   kb  = (u16*)carve((size_t)M * D * 2);        // K
  u16*   vt  = (u16*)carve((size_t)D * M * 2);        // V transposed [D, B*S]
  float* sc  = (float*)carve((size_t)B * S * S * 4);  // scores fp32
  u16*   pb  = (u16*)carve((size_t)B * S * S * 2);    // softmax probs bf16

  // 1) fp32 -> bf16 converts
  cvt_f32_bf16<<<dim3((unsigned)((M * E) / 1024)), 256, 0, stream>>>(x, xb, (int)(M * E));
  cvt_f32_bf16<<<dim3((unsigned)((D * E) / 1024)), 256, 0, stream>>>(Wq, wqb, D * E);
  cvt_f32_bf16<<<dim3((unsigned)((D * E) / 1024)), 256, 0, stream>>>(Wk, wkb, D * E);
  cvt_f32_bf16<<<dim3((unsigned)((D * E) / 1024)), 256, 0, stream>>>(Wv, wvb, D * E);

  dim3 blk(128);
  // 2) projections: [8192,1024] x [1024,1024]^T
  gemm_bf16_nt<false, false><<<dim3(D / 128, (unsigned)(M / 128), 1), blk, 0, stream>>>(
      xb, E, 0, wqb, E, 0, qb, D, 0, bq, 0.03125f /* 1/sqrt(1024) */, E);
  gemm_bf16_nt<false, false><<<dim3(D / 128, (unsigned)(M / 128), 1), blk, 0, stream>>>(
      xb, E, 0, wkb, E, 0, kb, D, 0, bk, 1.0f, E);
  gemm_bf16_nt<false, true><<<dim3(D / 128, (unsigned)(M / 128), 1), blk, 0, stream>>>(
      xb, E, 0, wvb, E, 0, vt, (int)M, 0, bv, 1.0f, E);

  // 3) scores[b] = q[b] * k[b]^T  (q already scaled), fp32 out
  gemm_bf16_nt<true, false><<<dim3(S / 128, S / 128, B), blk, 0, stream>>>(
      qb, D, (long long)S * D, kb, D, (long long)S * D,
      sc, S, (long long)S * S, nullptr, 1.0f, D);

  // 4) row softmax -> bf16 probs
  softmax_rows<<<dim3(B * S), 256, 0, stream>>>(sc, pb, S);

  // 5) out[b] = P[b] * V[b]  via Vt: C[s,d] = sum_t P[s,t] * Vt[d, b*S + t]
  gemm_bf16_nt<true, false><<<dim3(D / 128, S / 128, B), blk, 0, stream>>>(
      pb, S, (long long)S * S, vt, (int)M, (long long)S,
      out, D, (long long)S * D, nullptr, 1.0f, S);
}
